// SpGraphTransAttentionLayer_11304353923837
// MI455X (gfx1250) — matrix operand order repaired and checked by reading, not compiled, not run on hardware
//
#include <hip/hip_runtime.h>
#include <hip/hip_bf16.h>

typedef __bf16 bf16_t;
typedef __attribute__((ext_vector_type(16))) __bf16 v16bf;
typedef __attribute__((ext_vector_type(8)))  __bf16 v8bf;
typedef __attribute__((ext_vector_type(8)))  float  v8f;

#define GN 50000
#define GIN 512
#define GAD 256
#define GH 4
#define GDK 64
#define GE 500000
#define GCOLS 768        // q | k | v concatenated columns
#define RPB 32           // rows per block in the GEMM (last block: 16-row tail)

// ---- order-preserving float<->int key for atomicMax on signed int ----
__device__ __forceinline__ int fkey(float f) {
  int b = __float_as_int(f);
  return b >= 0 ? b : (b ^ 0x7fffffff);
}
__device__ __forceinline__ float funkey(int k) {
  return __int_as_float(k >= 0 ? k : (k ^ 0x7fffffff));
}

// ---- CDNA5 async global->LDS copy (ASYNCcnt-tracked), per ISA 10/15.18 ----
__device__ __forceinline__ void async_g2lds_b128(void* lds_ptr, const void* gptr) {
  unsigned ldsoff = (unsigned)(uintptr_t)lds_ptr;     // flat LDS addr low bits = LDS offset
  asm volatile("global_load_async_to_lds_b128 %0, %1, off"
               :: "v"(ldsoff), "v"((unsigned long long)(uintptr_t)gptr)
               : "memory");
}
__device__ __forceinline__ void wait_async_zero() {
  asm volatile("s_wait_asynccnt 0" ::: "memory");
}

#define WMMA_BF16(A, B, C) \
  __builtin_amdgcn_wmma_f32_16x16x32_bf16(false, (A), false, (B), (short)0, (C), false, false)

// ---------------------------------------------------------------------
// Kernel 1: build column-major bf16 hi/lo weight matrix wt[768][512]
// ---------------------------------------------------------------------
__global__ void gat_prep_w(const float* __restrict__ WQ,
                           const float* __restrict__ WK,
                           const float* __restrict__ WV,
                           bf16_t* __restrict__ wt_hi,
                           bf16_t* __restrict__ wt_lo) {
  int idx = blockIdx.x * 256 + threadIdx.x;
  if (idx >= GCOLS * GIN) return;
  int j  = idx / GIN;   // column 0..767
  int kk = idx % GIN;   // reduction index 0..511
  float w;
  if (j < GAD)            w = WQ[kk * GAD + j];
  else if (j < 2 * GAD)   w = WK[kk * GAD + (j - GAD)];
  else                    w = WV[kk * GAD + (j - 2 * GAD)];
  bf16_t hi = (bf16_t)w;
  wt_hi[idx] = hi;
  wt_lo[idx] = (bf16_t)(w - (float)hi);
}

// ---------------------------------------------------------------------
// Kernel 2: fused QKV projection, split-bf16 WMMA (hh + hl + lh)
// block = 32-row stripe (async-staged to LDS); 8 waves, each owning
// 6 column tiles x 2 row tiles with persistent accumulators; K-loop rolled.
// ---------------------------------------------------------------------
__global__ __launch_bounds__(256) void gat_gemm_qkv(
    const float* __restrict__ x,
    const bf16_t* __restrict__ wt_hi, const bf16_t* __restrict__ wt_lo,
    const float* __restrict__ bQ, const float* __restrict__ bK,
    const float* __restrict__ bV,
    float* __restrict__ qout, float* __restrict__ kout,
    float* __restrict__ vout) {
  __shared__ __align__(16) float  sStage[RPB * GIN];   // 64 KB f32 stage
  __shared__ __align__(32) bf16_t sAhi[RPB * GIN];     // 32 KB
  __shared__ __align__(32) bf16_t sAlo[RPB * GIN];     // 32 KB

  const int tid = threadIdx.x;
  const long rowbase = (long)blockIdx.x * RPB;
  const int rows_valid = (GN - rowbase) < RPB ? (int)(GN - rowbase) : RPB;
  const int validVec = rows_valid * GIN / 4;           // valid float4 count
  const float* xblk = x + rowbase * GIN;

  // Phase 1: async copy stripe f32 global -> LDS (no VGPR destinations)
  #pragma unroll
  for (int j = 0; j < (RPB * GIN / 4) / 256; ++j) {    // 16 issues/thread
    int i = j * 256 + tid;                             // float4 index
    if (i < validVec)
      async_g2lds_b128(&sStage[i * 4], xblk + (long)i * 4);
  }
  wait_async_zero();

  // Phase 2: split f32 -> bf16 hi/lo (zero-fill tail so WMMA reads are clean)
  #pragma unroll
  for (int j = 0; j < (RPB * GIN / 4) / 256; ++j) {
    int i = j * 256 + tid;
    int idx = i * 4;
    if (i < validVec) {
      float4 v = ((const float4*)sStage)[i];
      bf16_t h0 = (bf16_t)v.x, h1 = (bf16_t)v.y, h2 = (bf16_t)v.z, h3 = (bf16_t)v.w;
      sAhi[idx + 0] = h0; sAhi[idx + 1] = h1; sAhi[idx + 2] = h2; sAhi[idx + 3] = h3;
      sAlo[idx + 0] = (bf16_t)(v.x - (float)h0);
      sAlo[idx + 1] = (bf16_t)(v.y - (float)h1);
      sAlo[idx + 2] = (bf16_t)(v.z - (float)h2);
      sAlo[idx + 3] = (bf16_t)(v.w - (float)h3);
    } else {
      bf16_t z = (bf16_t)0.0f;
      sAhi[idx + 0] = z; sAhi[idx + 1] = z; sAhi[idx + 2] = z; sAhi[idx + 3] = z;
      sAlo[idx + 0] = z; sAlo[idx + 1] = z; sAlo[idx + 2] = z; sAlo[idx + 3] = z;
    }
  }
  __syncthreads();

  const int lane = tid & 31;
  const int wave = tid >> 5;
  // A operand (16-bit 16x32, ISA 7.12.2): lane -> row m, K sub-offset
  const int m0 = lane & 15;
  const int ka = (lane >> 4) * 16;
  // B operand: lane -> column n, K sub-offset
  const int nl = lane & 15;
  const int kb = (lane >> 4) * 16;
  // C/D: VGPR i -> M = mb + i, N = nl
  const int mb = (lane >> 4) * 8;

  // one base pointer per wave-lane; column tiles are +t*16*GIN elements
  const int col0 = wave * 96 + nl;                   // first of 6 columns
  const bf16_t* bh0 = wt_hi + (long)col0 * GIN + kb;
  const bf16_t* bl0 = wt_lo + (long)col0 * GIN + kb;

  v8f acc0[6] = {}, acc1[6] = {};                    // 12 x 16x16 f32 tiles

  #pragma unroll 1   // keep K-loop rolled: bounded live range, no spills
  for (int kt = 0; kt < 16; ++kt) {
    const int k0 = kt * 32;
    // A operands for both row tiles, loaded once per K-step, reused 6x
    v8bf a0 = *(const v8bf*)&sAhi[m0 * GIN + k0 + ka];
    v8bf a1 = *(const v8bf*)&sAhi[m0 * GIN + k0 + 8 + ka];
    v8bf l0 = *(const v8bf*)&sAlo[m0 * GIN + k0 + ka];
    v8bf l1 = *(const v8bf*)&sAlo[m0 * GIN + k0 + 8 + ka];
    v16bf ahi0 = __builtin_shufflevector(a0, a1, 0,1,2,3,4,5,6,7,8,9,10,11,12,13,14,15);
    v16bf alo0 = __builtin_shufflevector(l0, l1, 0,1,2,3,4,5,6,7,8,9,10,11,12,13,14,15);
    const int m1 = m0 + 16;
    v8bf b0 = *(const v8bf*)&sAhi[m1 * GIN + k0 + ka];
    v8bf b1 = *(const v8bf*)&sAhi[m1 * GIN + k0 + 8 + ka];
    v8bf n0 = *(const v8bf*)&sAlo[m1 * GIN + k0 + ka];
    v8bf n1 = *(const v8bf*)&sAlo[m1 * GIN + k0 + 8 + ka];
    v16bf ahi1 = __builtin_shufflevector(b0, b1, 0,1,2,3,4,5,6,7,8,9,10,11,12,13,14,15);
    v16bf alo1 = __builtin_shufflevector(n0, n1, 0,1,2,3,4,5,6,7,8,9,10,11,12,13,14,15);

    #pragma unroll
    for (int t = 0; t < 6; ++t) {
      const long boff = (long)t * 16 * GIN + k0;     // folds to immediate offsets
      v16bf bhi = *(const v16bf*)(bh0 + boff);
      v16bf blo = *(const v16bf*)(bl0 + boff);
      acc0[t] = WMMA_BF16(ahi0, bhi, acc0[t]);
      acc0[t] = WMMA_BF16(ahi0, blo, acc0[t]);
      acc0[t] = WMMA_BF16(alo0, bhi, acc0[t]);
      acc1[t] = WMMA_BF16(ahi1, bhi, acc1[t]);
      acc1[t] = WMMA_BF16(ahi1, blo, acc1[t]);
      acc1[t] = WMMA_BF16(alo1, bhi, acc1[t]);
    }
  }

  // bias + store all 12 tiles (branch uniform per wave; guard tail rows)
  #pragma unroll
  for (int t = 0; t < 6; ++t) {
    const int col = col0 + t * 16;       // global column 0..767
    #pragma unroll
    for (int rt = 0; rt < 2; ++rt) {
      const v8f& c = rt ? acc1[t] : acc0[t];
      const long rb = rowbase + rt * 16;
      if (col < GAD) {
        const float bias = bQ[col];
        #pragma unroll
        for (int i = 0; i < 8; ++i) {
          long r = rb + mb + i;
          if (r < GN) qout[r * GAD + col] = c[i] + bias;
        }
      } else if (col < 2 * GAD) {
        const int jj = col - GAD;
        const float bias = bK[jj];
        #pragma unroll
        for (int i = 0; i < 8; ++i) {
          long r = rb + mb + i;
          if (r < GN) kout[r * GAD + jj] = c[i] + bias;
        }
      } else {
        const int jj = col - 2 * GAD;       // V projection column
        const int h = jj >> 6, d = jj & 63; // v output is [N, d_k, h]
        const float bias = bV[jj];
        #pragma unroll
        for (int i = 0; i < 8; ++i) {
          long r = rb + mb + i;
          if (r < GN) vout[r * GAD + d * GH + h] = c[i] + bias;
        }
      }
    }
  }
}

// ---------------------------------------------------------------------
// Kernel 3: init segment max keys / sums
// ---------------------------------------------------------------------
__global__ void gat_seg_init(int* __restrict__ smax, float* __restrict__ ssum) {
  int i = blockIdx.x * 256 + threadIdx.x;
  if (i < GN * GH) {
    smax[i] = (int)0x80000000;  // <= fkey(any float)
    ssum[i] = 0.0f;
  }
}

// ---------------------------------------------------------------------
// Kernel 4: per-(edge,head) scaled dot product + segment max
// ---------------------------------------------------------------------
__global__ void gat_edge_logits(const int* __restrict__ e0,
                                const int* __restrict__ e1,
                                const float* __restrict__ q,
                                const float* __restrict__ k,
                                float* __restrict__ prods,
                                int* __restrict__ smax) {
  int t = blockIdx.x * 256 + threadIdx.x;
  if (t >= GE * GH) return;
  int e = t >> 2, h = t & 3;
  int s = e0[e], d = e1[e];
  const float4* qp = (const float4*)(q + (long)s * GAD + h * GDK);
  const float4* kp = (const float4*)(k + (long)d * GAD + h * GDK);
  float acc = 0.0f;
  #pragma unroll
  for (int i = 0; i < GDK / 4; ++i) {
    float4 a = qp[i], b = kp[i];
    acc += a.x * b.x + a.y * b.y + a.z * b.z + a.w * b.w;
  }
  float p = acc * 0.125f;  // 1/sqrt(64)
  prods[t] = p;
  atomicMax(&smax[s * GH + h], fkey(p));
}

// ---------------------------------------------------------------------
// Kernel 5: exp(p - segmax) + segment sum; writes unnormalized attention
// ---------------------------------------------------------------------
__global__ void gat_edge_exp(const int* __restrict__ e0,
                             const float* __restrict__ prods,
                             const int* __restrict__ smax,
                             float* __restrict__ ssum,
                             float* __restrict__ att) {
  int t = blockIdx.x * 256 + threadIdx.x;
  if (t >= GE * GH) return;
  int e = t >> 2, h = t & 3;
  int s = e0[e];
  float m = funkey(smax[s * GH + h]);
  float ex = __expf(prods[t] - m);
  att[t] = ex;
  atomicAdd(&ssum[s * GH + h], ex);
}

// ---------------------------------------------------------------------
// Kernel 6: normalize
// ---------------------------------------------------------------------
__global__ void gat_edge_norm(const int* __restrict__ e0,
                              const float* __restrict__ ssum,
                              float* __restrict__ att) {
  int t = blockIdx.x * 256 + threadIdx.x;
  if (t >= GE * GH) return;
  int e = t >> 2, h = t & 3;
  int s = e0[e];
  att[t] = att[t] / ssum[s * GH + h];
}

// ---------------------------------------------------------------------
extern "C" void kernel_launch(void* const* d_in, const int* in_sizes, int n_in,
                              void* d_out, int out_size, void* d_ws, size_t ws_size,
                              hipStream_t stream) {
  const float* x  = (const float*)d_in[0];
  const float* WQ = (const float*)d_in[1];
  const float* bQ = (const float*)d_in[2];
  const float* WK = (const float*)d_in[3];
  const float* bK = (const float*)d_in[4];
  const float* WV = (const float*)d_in[5];
  const float* bV = (const float*)d_in[6];
  const int*   e0 = (const int*)d_in[7];        // edge[0]
  const int*   e1 = ((const int*)d_in[7]) + GE; // edge[1]

  float* att  = (float*)d_out;                    // [E, H]
  float* vout = (float*)d_out + (size_t)GE * GH;  // [N, d_k, h]

  // workspace carve-up (256B aligned)
  char* ws = (char*)d_ws;
  size_t off = 0;
  auto take = [&](size_t bytes) {
    char* p = ws + off;
    off = (off + bytes + 255) & ~(size_t)255;
    return p;
  };
  bf16_t* wt_hi = (bf16_t*)take((size_t)GCOLS * GIN * sizeof(bf16_t));
  bf16_t* wt_lo = (bf16_t*)take((size_t)GCOLS * GIN * sizeof(bf16_t));
  float*  q     = (float*)take((size_t)GN * GAD * sizeof(float));
  float*  k     = (float*)take((size_t)GN * GAD * sizeof(float));
  float*  prods = (float*)take((size_t)GE * GH * sizeof(float));
  int*    smax  = (int*)take((size_t)GN * GH * sizeof(int));
  float*  ssum  = (float*)take((size_t)GN * GH * sizeof(float));
  (void)ws_size; (void)n_in; (void)in_sizes; (void)out_size;

  // 1) weight transpose + bf16 split
  gat_prep_w<<<(GCOLS * GIN + 255) / 256, 256, 0, stream>>>(WQ, WK, WV, wt_hi, wt_lo);

  // 2) fused QKV GEMM (1563 blocks; last one is a guarded 16-row tail)
  gat_gemm_qkv<<<(GN + RPB - 1) / RPB, 256, 0, stream>>>(x, wt_hi, wt_lo,
                                                         bQ, bK, bV, q, k, vout);

  // 3) segment state init
  gat_seg_init<<<(GN * GH + 255) / 256, 256, 0, stream>>>(smax, ssum);

  // 4-6) edge logits -> segment softmax
  const int eg = (GE * GH + 255) / 256;
  gat_edge_logits<<<eg, 256, 0, stream>>>(e0, e1, q, k, prods, smax);
  gat_edge_exp<<<eg, 256, 0, stream>>>(e0, prods, smax, ssum, att);
  gat_edge_norm<<<eg, 256, 0, stream>>>(e0, ssum, att);
}
